// Block_2989297238679
// MI455X (gfx1250) — matrix operand order repaired and checked by reading, not compiled, not run on hardware
//
#include <hip/hip_runtime.h>

// ---------------------------------------------------------------------------
// Types for CDNA5 WMMA (wave32)
// ---------------------------------------------------------------------------
typedef __bf16 bf16_t;
typedef __attribute__((ext_vector_type(16))) __bf16 v16bf;
typedef __attribute__((ext_vector_type(8)))  float  v8f;

#define Bb 2
#define Tt 2048
#define Ee 1024
#define Hh 16
#define HSz 64
#define Rr (Bb * Tt)   // 4096 token rows
#define FF (4 * Ee)    // 4096 ffn width

static __device__ __forceinline__ v8f wmma_bf16(v16bf a, v16bf b, v8f c) {
  // D = A(16x32 bf16) * B(32x16 bf16) + C(16x16 f32)
  return __builtin_amdgcn_wmma_f32_16x16x32_bf16(false, a, false, b, (short)0, c,
                                                 false, false);
}

static __device__ __forceinline__ float half_max(float v) {
  v = fmaxf(v, __shfl_xor(v, 1, 32));
  v = fmaxf(v, __shfl_xor(v, 2, 32));
  v = fmaxf(v, __shfl_xor(v, 4, 32));
  v = fmaxf(v, __shfl_xor(v, 8, 32));
  return v;
}
static __device__ __forceinline__ float half_sum(float v) {
  v += __shfl_xor(v, 1, 32);
  v += __shfl_xor(v, 2, 32);
  v += __shfl_xor(v, 4, 32);
  v += __shfl_xor(v, 8, 32);
  return v;
}

// CDNA5 async memory->LDS DMA (ASYNCcnt-tracked, bypasses VGPRs).
// lds: byte offset within workgroup LDS (low 32 bits of generic address).
// off: per-lane unsigned byte offset; base: uniform 64-bit global base (SGPR pair).
static __device__ __forceinline__ void async_ld_b128(uint32_t lds, uint32_t off,
                                                     const void* base) {
  asm volatile("global_load_async_to_lds_b128 %0, %1, %2"
               :: "v"(lds), "v"(off), "s"(base) : "memory");
}
static __device__ __forceinline__ void wait_async0() {
  asm volatile("s_wait_asynccnt 0" ::: "memory");
}
static __device__ __forceinline__ uint32_t lds_addr(const void* p) {
  return (uint32_t)(size_t)p;
}

// ---------------------------------------------------------------------------
// Conversion / packing kernels (one-time, tiny vs GEMM cost)
// ---------------------------------------------------------------------------
__global__ void f32_to_bf16_kernel(const float* __restrict__ in,
                                   bf16_t* __restrict__ out, int n) {
  int i = blockIdx.x * blockDim.x + threadIdx.x;
  if (i < n) out[i] = (bf16_t)in[i];
}

// in: [Kd][Nd] f32 row-major -> out: [Nd][Kd] bf16 (transposed for async staging)
__global__ void f32_to_bf16_T_kernel(const float* __restrict__ in,
                                     bf16_t* __restrict__ out, int Kd, int Nd) {
  int i = blockIdx.x * blockDim.x + threadIdx.x;
  if (i >= Kd * Nd) return;
  int n = i / Kd, k = i % Kd;
  out[i] = (bf16_t)in[(size_t)k * Nd + n];
}

// Wq/Wk/Wv: [H, E, HS] each -> packed TRANSPOSED bf16 [3*E, E]:
// row c = (q|k|v, h*64+d), col e.
__global__ void pack_qkv_kernel(const float* __restrict__ Wq,
                                const float* __restrict__ Wk,
                                const float* __restrict__ Wv,
                                bf16_t* __restrict__ Wqkv) {
  int i = blockIdx.x * blockDim.x + threadIdx.x;
  if (i >= 3 * Ee * Ee) return;
  int c = i >> 10, e = i & 1023;      // out[c][e], Ee == 1024
  int sel = c >> 10, cc = c & 1023;
  int h = cc >> 6, d = cc & 63;
  const float* W = (sel == 0) ? Wq : (sel == 1) ? Wk : Wv;
  Wqkv[i] = (bf16_t)W[((size_t)(h * Ee + e)) * HSz + d];
}

// ---------------------------------------------------------------------------
// Tiled bf16 WMMA GEMM: C = A(MxK) * B(KxN) [+bias][+relu], B given TRANSPOSED
// as Bt[N][K]. block = 256 (8 waves), tile 128x128, K staged 32-deep with
// double-buffered async DMA into LDS (overlaps with WMMA compute).
// MODE 0: f32 out + bias   MODE 1: bf16 out + bias + relu   MODE 2: bf16 out
// ---------------------------------------------------------------------------
template <int MODE>
__global__ void __launch_bounds__(256)
gemm_bf16_kernel(const bf16_t* __restrict__ A, const bf16_t* __restrict__ Bt,
                 const float* __restrict__ bias, void* __restrict__ Cv,
                 int M, int N, int K) {
  __shared__ alignas(16) bf16_t As[2][128][40];   // [m][k] padded (80B rows)
  __shared__ alignas(16) bf16_t Bs[2][128][40];   // [n][k] padded
  const int tid  = threadIdx.x;
  const int w    = tid >> 5;
  const int l    = tid & 31;
  const int half = l >> 4;
  const int ln   = l & 15;
  const int row0 = blockIdx.y * 128;
  const int n0   = blockIdx.x * 128;
  const int NK   = K >> 5;

  // issue one 128x32 A tile + 128x32 B tile as 16B async chunks (4 per thread)
  auto issue = [&](int buf, int k0) {
    const bf16_t* Abase = A  + (size_t)row0 * K + k0;
    const bf16_t* Bbase = Bt + (size_t)n0   * K + k0;
#pragma unroll
    for (int i = 0; i < 2; ++i) {
      int c = tid + i * 256;                    // 512 chunks of 16B
      int m = c >> 2, kc = c & 3;
      async_ld_b128(lds_addr(&As[buf][m][kc * 8]),
                    (uint32_t)((m * K + kc * 8) * 2), Abase);
    }
#pragma unroll
    for (int i = 0; i < 2; ++i) {
      int c = tid + i * 256;
      int n = c >> 2, kc = c & 3;
      async_ld_b128(lds_addr(&Bs[buf][n][kc * 8]),
                    (uint32_t)((n * K + kc * 8) * 2), Bbase);
    }
  };

  v8f acc[8];
#pragma unroll
  for (int i = 0; i < 8; ++i) acc[i] = (v8f){0.f, 0.f, 0.f, 0.f, 0.f, 0.f, 0.f, 0.f};

  issue(0, 0);
  wait_async0();
  __syncthreads();

  for (int kt = 0; kt < NK; ++kt) {
    const int cur = kt & 1;
    if (kt + 1 < NK) issue(cur ^ 1, (kt + 1) * 32);   // DMA next while computing

    v16bf af;                                   // A fragment: lane row = ln
    {
      const bf16_t* p = &As[cur][w * 16 + ln][0];
#pragma unroll
      for (int j = 0; j < 8; ++j) af[j] = p[half * 16 + j];
#pragma unroll
      for (int j = 0; j < 8; ++j) af[8 + j] = p[half * 16 + 8 + j];
    }
#pragma unroll
    for (int nt = 0; nt < 8; ++nt) {            // B fragment: lane col = ln
      v16bf bfr;
      const bf16_t* p = &Bs[cur][nt * 16 + ln][half * 16];
#pragma unroll
      for (int j = 0; j < 16; ++j) bfr[j] = p[j];
      acc[nt] = wmma_bf16(af, bfr, acc[nt]);
    }
    wait_async0();       // next tile fully in LDS (our wave's DMA)
    __syncthreads();     // everyone's DMA visible; frees buffer we just read
  }

  // epilogue; C layout: row = r + 8*half, col = ln
  const int rowb = row0 + w * 16 + half * 8;
#pragma unroll
  for (int nt = 0; nt < 8; ++nt) {
    int col = n0 + nt * 16 + ln;
    float bv = 0.f;
    if (MODE != 2) bv = bias[col];
#pragma unroll
    for (int r = 0; r < 8; ++r) {
      size_t idx = (size_t)(rowb + r) * N + col;
      float v = acc[nt][r] + bv;
      if (MODE == 0) {
        ((float*)Cv)[idx] = v;
      } else if (MODE == 1) {
        ((bf16_t*)Cv)[idx] = (bf16_t)fmaxf(v, 0.f);
      } else {
        ((bf16_t*)Cv)[idx] = (bf16_t)v;
      }
    }
  }
}

// ---------------------------------------------------------------------------
// Flash attention (causal, online softmax), bf16 WMMA, f32 accumulation.
// grid = (T/128, B*H); block = 256 (8 waves); wave -> 16 query rows.
// K/V 32x64 chunks double-buffered via async DMA into LDS.
// qkv: [B*T, 3072] bf16 (q | k | v). out: [B*T, E] bf16 (heads concatenated).
// ---------------------------------------------------------------------------
__global__ void __launch_bounds__(256)
attn_kernel(const bf16_t* __restrict__ qkv, bf16_t* __restrict__ attnb) {
  __shared__ alignas(16) bf16_t Ks[2][32][72];   // [s][d] padded (144B rows)
  __shared__ alignas(16) bf16_t Vs[2][32][72];
  __shared__ alignas(16) bf16_t Ps[8][16][40];   // per-wave softmax tile
  const int tid  = threadIdx.x;
  const int w    = tid >> 5;
  const int l    = tid & 31;
  const int half = l >> 4;
  const int ln   = l & 15;
  const int bh   = blockIdx.y;
  const int b    = bh >> 4;            // H == 16
  const int h    = bh & 15;
  const int qbase = blockIdx.x * 128;
  const size_t rowB = (size_t)b * Tt;
  const float scale = 0.125f;          // 64^-0.5

  // issue one 32x64 K chunk + V chunk (16B async chunks, 2 per thread)
  auto issue_kv = [&](int buf, int s0) {
    int r = tid >> 3, cc = tid & 7;                  // 256 chunks per tile
    uint32_t go = (uint32_t)((r * 3 * Ee + cc * 8) * 2);
    const bf16_t* kb = qkv + (rowB + s0) * (size_t)(3 * Ee) + Ee + h * 64;
    const bf16_t* vb = qkv + (rowB + s0) * (size_t)(3 * Ee) + 2 * Ee + h * 64;
    async_ld_b128(lds_addr(&Ks[buf][r][cc * 8]), go, kb);
    async_ld_b128(lds_addr(&Vs[buf][r][cc * 8]), go, vb);
  };

  // Q fragments for this wave's 16 rows (d 0..31 and 32..63)
  v16bf aq0, aq1;
  {
    const bf16_t* qp = qkv + (rowB + qbase + w * 16 + ln) * (size_t)(3 * Ee) + h * 64;
#pragma unroll
    for (int j = 0; j < 8; ++j) {
      aq0[j]     = qp[half * 16 + j];
      aq0[8 + j] = qp[half * 16 + 8 + j];
      aq1[j]     = qp[32 + half * 16 + j];
      aq1[8 + j] = qp[40 + half * 16 + j];
    }
  }

  v8f o[4];
#pragma unroll
  for (int i = 0; i < 4; ++i) o[i] = (v8f){0.f, 0.f, 0.f, 0.f, 0.f, 0.f, 0.f, 0.f};
  float m_run[8], l_run[8];
#pragma unroll
  for (int r = 0; r < 8; ++r) { m_run[r] = -1e30f; l_run[r] = 0.f; }

  const int qw = qbase + w * 16;
  const int NC = (qbase >> 5) + 4;     // key chunks, uniform for whole block

  issue_kv(0, 0);
  wait_async0();
  __syncthreads();

  for (int kt = 0; kt < NC; ++kt) {
    const int s0  = kt * 32;
    const int cur = kt & 1;
    if (kt + 1 < NC) issue_kv(cur ^ 1, (kt + 1) * 32);   // prefetch next chunk

    // scores S(16q x 32k) = Q * K^T : two 16x16 C tiles, K-dim = d (2 steps)
    v8f c0 = (v8f){0.f, 0.f, 0.f, 0.f, 0.f, 0.f, 0.f, 0.f};
    v8f c1 = c0;
    {
      v16bf bk;
      const bf16_t* p;
      p = &Ks[cur][ln][half * 16];
#pragma unroll
      for (int j = 0; j < 16; ++j) bk[j] = p[j];
      c0 = wmma_bf16(aq0, bk, c0);
      p = &Ks[cur][ln][32 + half * 16];
#pragma unroll
      for (int j = 0; j < 16; ++j) bk[j] = p[j];
      c0 = wmma_bf16(aq1, bk, c0);
      p = &Ks[cur][16 + ln][half * 16];
#pragma unroll
      for (int j = 0; j < 16; ++j) bk[j] = p[j];
      c1 = wmma_bf16(aq0, bk, c1);
      p = &Ks[cur][16 + ln][32 + half * 16];
#pragma unroll
      for (int j = 0; j < 16; ++j) bk[j] = p[j];
      c1 = wmma_bf16(aq1, bk, c1);
    }

    // online softmax per row (row = half*8 + r, lives in one 16-lane half)
    float al[8];
#pragma unroll
    for (int r = 0; r < 8; ++r) {
      int qg  = qw + half * 8 + r;
      int kg0 = s0 + ln;
      int kg1 = s0 + 16 + ln;
      float s0v = c0[r] * scale; if (kg0 > qg) s0v = -1e30f;
      float s1v = c1[r] * scale; if (kg1 > qg) s1v = -1e30f;
      float mx = half_max(fmaxf(s0v, s1v));
      float mnew  = fmaxf(m_run[r], mx);
      float alpha = __expf(m_run[r] - mnew);
      float p0 = __expf(s0v - mnew);
      float p1 = __expf(s1v - mnew);
      l_run[r] = l_run[r] * alpha + half_sum(p0 + p1);
      m_run[r] = mnew;
      al[r] = alpha;
      int prow = half * 8 + r;
      Ps[w][prow][ln]      = (bf16_t)p0;
      Ps[w][prow][16 + ln] = (bf16_t)p1;
    }
#pragma unroll
    for (int r = 0; r < 8; ++r) {
      o[0][r] *= al[r]; o[1][r] *= al[r]; o[2][r] *= al[r]; o[3][r] *= al[r];
    }

    // wave-private LDS bounce: make P visible to sibling lanes
    asm volatile("s_wait_dscnt 0" ::: "memory");

    v16bf pa;                           // P as A-fragment (16x32)
    {
      const bf16_t* p = &Ps[w][ln][0];
#pragma unroll
      for (int j = 0; j < 8; ++j) pa[j] = p[half * 16 + j];
#pragma unroll
      for (int j = 0; j < 8; ++j) pa[8 + j] = p[half * 16 + 8 + j];
    }
    // O += P(16x32) * V(32x64): 4 N-tiles; B fragment K-dim = key index
#pragma unroll
    for (int nt = 0; nt < 4; ++nt) {
      v16bf bv;
      const bf16_t* p = &Vs[cur][half * 16][nt * 16 + ln];
#pragma unroll
      for (int j = 0; j < 16; ++j) bv[j] = p[j * 72];
      o[nt] = wmma_bf16(pa, bv, o[nt]);
    }
    wait_async0();
    __syncthreads();
  }

  // finalize: O / l, store into concat-head layout [B*T, E] at col h*64+...
#pragma unroll
  for (int r = 0; r < 8; ++r) {
    int qg = qw + half * 8 + r;
    float inv = 1.0f / l_run[r];
#pragma unroll
    for (int nt = 0; nt < 4; ++nt) {
      attnb[(rowB + qg) * (size_t)Ee + h * 64 + nt * 16 + ln] =
          (bf16_t)(o[nt][r] * inv);
    }
  }
}

// ---------------------------------------------------------------------------
// y = LayerNorm(a + res) * g + be ; optional bf16 mirror for next GEMM
// ---------------------------------------------------------------------------
__global__ void __launch_bounds__(256)
add_ln_kernel(const float* __restrict__ a, const float* __restrict__ res,
              const float* __restrict__ g, const float* __restrict__ be,
              float* __restrict__ y, bf16_t* __restrict__ yb) {
  __shared__ float s1[256], s2[256];
  const int row = blockIdx.x;
  const int tid = threadIdx.x;
  const float* pa = a + (size_t)row * Ee;
  const float* pr = res + (size_t)row * Ee;
  float sum = 0.f, sq = 0.f;
  for (int i = tid; i < Ee; i += 256) {
    float v = pa[i] + pr[i];
    sum += v; sq += v * v;
  }
  s1[tid] = sum; s2[tid] = sq;
  __syncthreads();
  for (int st = 128; st > 0; st >>= 1) {
    if (tid < st) { s1[tid] += s1[tid + st]; s2[tid] += s2[tid + st]; }
    __syncthreads();
  }
  float mu  = s1[0] * (1.0f / Ee);
  float var = s2[0] * (1.0f / Ee) - mu * mu;
  float rs  = rsqrtf(var + 1e-5f);
  for (int i = tid; i < Ee; i += 256) {
    float v = (pa[i] + pr[i] - mu) * rs * g[i] + be[i];
    y[(size_t)row * Ee + i] = v;
    if (yb) yb[(size_t)row * Ee + i] = (bf16_t)v;
  }
}

// ---------------------------------------------------------------------------
// Host-side orchestration
// ---------------------------------------------------------------------------
extern "C" void kernel_launch(void* const* d_in, const int* in_sizes, int n_in,
                              void* d_out, int out_size, void* d_ws, size_t ws_size,
                              hipStream_t stream) {
  (void)in_sizes; (void)n_in; (void)out_size; (void)ws_size;
  const float* x   = (const float*)d_in[0];
  const float* Wq  = (const float*)d_in[1];
  const float* Wk  = (const float*)d_in[2];
  const float* Wv  = (const float*)d_in[3];
  const float* Wo  = (const float*)d_in[4];
  const float* bo  = (const float*)d_in[5];
  const float* W1  = (const float*)d_in[6];
  const float* b1  = (const float*)d_in[7];
  const float* W2  = (const float*)d_in[8];
  const float* b2  = (const float*)d_in[9];
  const float* g1  = (const float*)d_in[10];
  const float* be1 = (const float*)d_in[11];
  const float* g2  = (const float*)d_in[12];
  const float* be2 = (const float*)d_in[13];
  float* out = (float*)d_out;

  char* ws = (char*)d_ws;
  size_t off = 0;
  auto carve = [&](size_t bytes) -> void* {
    void* p = ws + off;
    off += (bytes + 255) & ~(size_t)255;
    return p;
  };
  bf16_t* xb    = (bf16_t*)carve((size_t)Rr * Ee * 2);         // A operands
  bf16_t* Wqkvb = (bf16_t*)carve((size_t)3 * Ee * Ee * 2);     // [3E][E] (T)
  bf16_t* Wob   = (bf16_t*)carve((size_t)Ee * Ee * 2);         // [E][E]  (T)
  bf16_t* W1b   = (bf16_t*)carve((size_t)FF * Ee * 2);         // [FF][E] (T)
  bf16_t* W2b   = (bf16_t*)carve((size_t)Ee * FF * 2);         // [E][FF] (T)
  bf16_t* qkvb  = (bf16_t*)carve((size_t)Rr * 3 * Ee * 2);
  bf16_t* attnb = (bf16_t*)carve((size_t)Rr * Ee * 2);
  float*  proj  = (float*)carve((size_t)Rr * Ee * 4);          // reused as ff
  float*  y1    = (float*)carve((size_t)Rr * Ee * 4);
  bf16_t* y1b   = (bf16_t*)carve((size_t)Rr * Ee * 2);
  bf16_t* hb    = (bf16_t*)carve((size_t)Rr * FF * 2);

  int n;
  n = Rr * Ee;
  f32_to_bf16_kernel<<<(n + 255) / 256, 256, 0, stream>>>(x, xb, n);
  n = 3 * Ee * Ee;
  pack_qkv_kernel<<<(n + 255) / 256, 256, 0, stream>>>(Wq, Wk, Wv, Wqkvb);
  n = Ee * Ee;
  f32_to_bf16_T_kernel<<<(n + 255) / 256, 256, 0, stream>>>(Wo, Wob, Ee, Ee);
  n = Ee * FF;
  f32_to_bf16_T_kernel<<<(n + 255) / 256, 256, 0, stream>>>(W1, W1b, Ee, FF);
  n = FF * Ee;
  f32_to_bf16_T_kernel<<<(n + 255) / 256, 256, 0, stream>>>(W2, W2b, FF, Ee);

  // qkv = x @ [Wq|Wk|Wv]   (4096 x 3072, K=1024), bf16 out
  gemm_bf16_kernel<2><<<dim3(3 * Ee / 128, Rr / 128), 256, 0, stream>>>(
      xb, Wqkvb, nullptr, qkvb, Rr, 3 * Ee, Ee);

  // causal flash attention -> concat heads [4096, 1024] bf16
  attn_kernel<<<dim3(Tt / 128, Bb * Hh), 256, 0, stream>>>(qkvb, attnb);

  // proj = attn @ Wo + bo   (f32 out)
  gemm_bf16_kernel<0><<<dim3(Ee / 128, Rr / 128), 256, 0, stream>>>(
      attnb, Wob, bo, proj, Rr, Ee, Ee);

  // y1 = LN1(x + proj)  (f32 + bf16 mirror)
  add_ln_kernel<<<Rr, 256, 0, stream>>>(proj, x, g1, be1, y1, y1b);

  // h = relu(y1 @ W1 + b1)  (bf16 out)
  gemm_bf16_kernel<1><<<dim3(FF / 128, Rr / 128), 256, 0, stream>>>(
      y1b, W1b, b1, hb, Rr, FF, Ee);

  // ff = h @ W2 + b2  (f32, reuse proj buffer)
  gemm_bf16_kernel<0><<<dim3(Ee / 128, Rr / 128), 256, 0, stream>>>(
      hb, W2b, b2, proj, Rr, Ee, FF);

  // out = LN2(y1 + ff)
  add_ln_kernel<<<Rr, 256, 0, stream>>>(proj, y1, g2, be2, out, nullptr);
}
